// TimeVarFIRFilter_29575144800679
// MI455X (gfx1250) — matrix operand order repaired and checked by reading, not compiled, not run on hardware
//
#include <hip/hip_runtime.h>

typedef __attribute__((ext_vector_type(16))) _Float16 v16h;
typedef __attribute__((ext_vector_type(8)))  float    v8f;
typedef __attribute__((ext_vector_type(4)))  float    v4f;

#define BB   16
#define TT   32768
#define KK   128

// Scheduling fence: nothing may cross (blocks MI-sched from moving DS/VMEM
// ops across our asynccnt waits / async-copy issues).
#if __has_builtin(__builtin_amdgcn_sched_barrier)
#define SCHED_FENCE() __builtin_amdgcn_sched_barrier(0)
#else
#define SCHED_FENCE() asm volatile("" ::: "memory")
#endif

// ---------------- LDS layout (byte offsets from start of the single
// __shared__ allocation, which sits at LDS offset 0; async-to-LDS VDST
// operands are byte offsets into the workgroup's LDS allocation) ----------
#define SIG_OFF      0        // 256 floats (need 255)          -> 1024 B
#define FBUF_OFF     1536     // 8 waves * 2 bufs * 2560 B      -> 40960 B
#define FBUF_STRIDE  2560     // one 16-row chunk buffer
#define ROW_STRIDE_B 144      // 36 floats/row: 32 data + pad (conflict-free)
#define CBUF_OFF     42496    // 8 waves * 16*16 f32            -> 8192 B
#define LDS_TOTAL    50688

__global__ __launch_bounds__(256)
void fir_wmma_kernel(const float* __restrict__ signal,
                     const float* __restrict__ fcoef,
                     float* __restrict__ out)
{
    __shared__ __align__(16) unsigned char smem[LDS_TOTAL];
    float* sig = (float*)(smem + SIG_OFF);

    const int tid  = threadIdx.x;
    const int lane = tid & 31;
    const int w    = tid >> 5;                 // wave id, 0..7
    const int tilesPerB = TT / 128;            // 256 blocks per batch row
    const int b        = blockIdx.x / tilesPerB;
    const int t0_block = (blockIdx.x % tilesPerB) * 128;
    const int t0       = t0_block + w * 16;    // this wave's 16 outputs

    // ---- stage the signal window [t0_block-127 .. t0_block+127] ----------
    {
        int g   = t0_block - 127 + tid;        // tid = 0..255 covers 255+1
        float v = 0.0f;
        if (g >= 0 && g < TT) v = signal[(size_t)b * TT + g];
        sig[tid] = v;                          // zero-pad out-of-range
    }
    __syncthreads();

    const float* fbase = fcoef + (size_t)b * TT * KK;   // SGPR-pair base

    // Async-copy one 16t x 32k f32 chunk (2 KB) into this wave's LDS buffer.
    // 4 x global_load_async_to_lds_b128: lane m=q*32+lane -> row i = m>>3,
    // 16-byte column (m&7)*16. Rows land at a 144 B stride (bank padding).
    auto issue_chunk = [&](int k0, int buf) {
        SCHED_FENCE();   // keep older reads of the target buffer above this
#pragma unroll
        for (int q = 0; q < 4; ++q) {
            int m    = q * 32 + lane;
            int i    = m >> 3;
            int colb = (m & 7) * 16;
            int voff = (t0 + i) * (KK * 4) + k0 * 4 + colb;      // global byte off
            int loff = FBUF_OFF + (w * 2 + buf) * FBUF_STRIDE
                     + i * ROW_STRIDE_B + colb;                  // LDS byte off
            asm volatile("global_load_async_to_lds_b128 %0, %1, %2"
                         :: "v"(loff), "v"(voff), "s"(fbase) : "memory");
        }
    };

    v8f acc = {};
    issue_chunk(0, 0);

#pragma unroll
    for (int c = 0; c < 4; ++c) {              // K chunks of 32
        const int buf = c & 1;
        if (c < 3) {
            issue_chunk((c + 1) * 32, buf ^ 1);
            // 4 newest ops outstanding belong to chunk c+1; <=4 means all of
            // chunk c has retired (async loads complete in order).
            asm volatile("s_wait_asynccnt 4" ::: "memory");
        } else {
            asm volatile("s_wait_asynccnt 0" ::: "memory");
        }
        SCHED_FENCE();   // no DS read of this buffer may hoist above the wait

        const float* fb = (const float*)(smem + FBUF_OFF
                                        + (w * 2 + buf) * FBUF_STRIDE);
        const int hi   = lane >> 4;
        const int mrow = lane & 15;

        // ---- A: shifted signal windows, f32 -> f16 -------------------
        // element e -> K = (e<8 ? e : e+8) + 8*hi ; value = sig[S + ...]
        const int S = w * 16 + mrow + c * 32 + hi * 8;
        v16h a;
#pragma unroll
        for (int e = 0; e < 8; ++e) a[e]     = (_Float16)sig[S + e];
#pragma unroll
        for (int e = 0; e < 8; ++e) a[8 + e] = (_Float16)sig[S + 16 + e];

        // ---- B: f_coef chunk transposed: element e -> K=(e<8 ? e : e+8)+8*hi, N=mrow
        v16h bm;
        const float* rowp = fb + mrow * 36 + hi * 8;
#pragma unroll
        for (int h = 0; h < 4; ++h) {
            v4f f4 = *(const v4f*)(rowp + h * 4 + (h >> 1) * 8);      // 16B-aligned LDS load
            bm[h * 4 + 0] = (_Float16)f4.x;
            bm[h * 4 + 1] = (_Float16)f4.y;
            bm[h * 4 + 2] = (_Float16)f4.z;
            bm[h * 4 + 3] = (_Float16)f4.w;
        }

        acc = __builtin_amdgcn_wmma_f32_16x16x32_f16(
                  false, a, false, bm, (short)0, acc, false, false);
    }

    // ---- spill C tile to LDS, read diagonal, store --------------------
    float* cb = (float*)(smem + CBUF_OFF) + w * 256;
    {
        const int hi = lane >> 4;
        const int n  = lane & 15;
#pragma unroll
        for (int r = 0; r < 8; ++r)
            cb[(r + 8 * hi) * 16 + n] = acc[r];
    }
    asm volatile("s_wait_dscnt 0" ::: "memory");   // same-wave LDS ordering
    SCHED_FENCE();

    if (lane < 16) {
        const int t = t0 + lane;
        out[(size_t)b * TT + t] = cb[lane * 16 + lane];
    }
}

extern "C" void kernel_launch(void* const* d_in, const int* in_sizes, int n_in,
                              void* d_out, int out_size, void* d_ws, size_t ws_size,
                              hipStream_t stream) {
    const float* signal = (const float*)d_in[0];   // (B, T)    f32
    const float* fcoef  = (const float*)d_in[1];   // (B, T, K) f32
    float* out = (float*)d_out;                    // (B, 1, T) f32

    dim3 grid(BB * (TT / 128));   // 4096 blocks, one b-row 128-t tile each
    dim3 block(256);              // 8 wave32s, one 16x16 WMMA tile per wave
    fir_wmma_kernel<<<grid, block, 0, stream>>>(signal, fcoef, out);
}
